// MLP_SAT_27333171872628
// MI455X (gfx1250) — compile-verified, matrix operand order checked by reading, not run on hardware
//
#include <hip/hip_runtime.h>
#include <hip/hip_bf16.h>

// CDNA5 / gfx1250 edge-MLP + segment-mean, v2: register-level B reuse.
//
// Block = 128 threads (4 wave32) = 64 edges.
//  Stage:  gather 64x768 fp32 -> bf16 WMMA-A tiles in LDS (96 KB).
//  L1:     wave w owns N-tiles {4w..4w+3} x all 4 M-tiles.
//          acc = 4x4 v8f (128 VGPRs); per K-step each B frag feeds 4 WMMAs,
//          each A frag (from LDS) feeds 4 WMMAs.  24 K-steps.
//  L2:     wave w owns N-tile w x all 4 M-tiles; 8 K-steps; B reuse x4.
//  Tail:   64->20->1 on VALU (threads 0..63), atomic segment accumulate.
//
// Workspace layout (needs ~1.2 MB):
//   [0)       W1 bf16 tiles : 393216 B
//   [393216)  W2 bf16 tiles :  32768 B
//   [425984)  sums   f32    : 262144 B
//   [688128)  counts f32    : 262144 B
//   [950272)  means  f32    : 262144 B

typedef __attribute__((ext_vector_type(16))) __bf16 v16bf;
typedef __attribute__((ext_vector_type(8)))  __bf16 v8bf;
typedef __attribute__((ext_vector_type(8)))  float  v8f;

#define N_CLAUSES 65536
#define N_EDGES   524288
#define FEAT      384
#define K1        768
#define N1        256
#define K2        256
#define N2        64

#define BLK_EDGES 64                  // edges per block (4 M-tiles)
#define LDS_A1_BYTES (4 * 24 * 512 * 2)   //  98304: [mt][kt][lane][16] bf16
#define LDS_H1_OFF   LDS_A1_BYTES         //  h1: [mt][kt2][lane][16] bf16 = 32768
#define LDS_H2_OFF   (LDS_A1_BYTES + 4 * 8 * 512 * 2)  // h2: 64x64 f32 = 16384
#define LDS_TOTAL    (LDS_H2_OFF + 64 * 64 * 4)        // 147456 B

// ---------------------------------------------------------------------------
// Repack row-major fp32 [K][N] weights into bf16 WMMA-B tiles (ISA 7.12.2):
// lane = n%16 + 16*((k%16)/8) ; e = 8*((k%32)/16) + (k%8); 512 bf16 per tile.
// ---------------------------------------------------------------------------
__global__ void prep_wmma_b_kernel(const float* __restrict__ W,
                                   __bf16* __restrict__ out, int K, int N) {
  int idx = blockIdx.x * blockDim.x + threadIdx.x;
  if (idx >= K * N) return;
  int k = idx / N, n = idx % N;
  int kt = k >> 5, kk = k & 31, nt = n >> 4, nn = n & 15;
  int lane = nn + (((kk >> 3) & 1) << 4);
  int e    = ((kk >> 4) << 3) | (kk & 7);
  int ntiles_n = N >> 4;
  out[((((kt * ntiles_n + nt) * 32) + lane) << 4) + e] = (__bf16)W[idx];
}

__global__ void zero_f32_kernel(float* __restrict__ p, int n) {
  int i = blockIdx.x * blockDim.x + threadIdx.x;
  if (i < n) p[i] = 0.0f;
}

// ---------------------------------------------------------------------------
__global__ __launch_bounds__(128) void edge_mlp_kernel(
    const float* __restrict__ l_embs, const float* __restrict__ c_embs,
    const int* __restrict__ edge_i, const int* __restrict__ edge_j,
    const __bf16* __restrict__ W1b, const float* __restrict__ b1,
    const __bf16* __restrict__ W2b, const float* __restrict__ b2,
    const float* __restrict__ W3,  const float* __restrict__ b3,
    const float* __restrict__ W4,  const float* __restrict__ b4,
    float* __restrict__ sums, float* __restrict__ cnts) {
  extern __shared__ __align__(32) char smem[];
  __bf16* A1 = (__bf16*)smem;                    // staged layer-1 A tiles
  __bf16* h1 = (__bf16*)(smem + LDS_H1_OFF);     // layer-2 A tiles
  float*  h2 = (float*)(smem + LDS_H2_OFF);      // layer-3 input, [64][64] f32

  const int tid  = threadIdx.x;
  const int lane = tid & 31;
  const int wave = tid >> 5;           // 0..3: owns N-tile strip
  const int m    = lane & 15;
  const int half = lane >> 4;
  const int base_edge = blockIdx.x * BLK_EDGES;

  // ---------------- Stage: gather -> bf16 A-layout tiles in LDS -----------
  {
    const int me = tid >> 1;           // edge-in-block 0..63
    const int h  = tid & 1;            // which 16-feature half of a K-tile
    const int ml = me & 15, mt = me >> 4;
    const int li = edge_i[base_edge + me];
    const int cj = edge_j[base_edge + me];
    const float* lrow = l_embs + (size_t)li * FEAT;
    const float* crow = c_embs + (size_t)cj * FEAT;
    for (int kt = 0; kt < 24; ++kt) {
      const int k0 = kt * 32;
      const float* rowp = (k0 < FEAT) ? (lrow + k0) : (crow + (k0 - FEAT));
      const float4* p = (const float4*)(rowp + h * 16);
      float buf[16];
      *(float4*)&buf[0]  = p[0];
      *(float4*)&buf[4]  = p[1];
      *(float4*)&buf[8]  = p[2];
      *(float4*)&buf[12] = p[3];
      v8bf r0, r1;
#pragma unroll
      for (int t = 0; t < 8; ++t) {
        r0[t] = (__bf16)buf[t];
        r1[t] = (__bf16)buf[8 + t];
      }
      const int toff = (mt * 24 + kt) * 512;
      *(v8bf*)&A1[toff + ml * 16 + h * 8]        = r0;   // lanes 0..15 run
      *(v8bf*)&A1[toff + (ml + 16) * 16 + h * 8] = r1;   // lanes 16..31 run
    }
  }
  __syncthreads();

  // ---------------- Layer 1: 768 -> 256 (wave strip: nt = 4w..4w+3) -------
  v8f acc[4][4];                        // [mt][p]
#pragma unroll
  for (int p = 0; p < 4; ++p) {
    float bv = b1[(wave * 4 + p) * 16 + m];
    v8f t;
#pragma unroll
    for (int r = 0; r < 8; ++r) t[r] = bv;
#pragma unroll
    for (int mt = 0; mt < 4; ++mt) acc[mt][p] = t;
  }

  for (int kt = 0; kt < 24; ++kt) {
    v16bf a[4];
#pragma unroll
    for (int mt = 0; mt < 4; ++mt)
      a[mt] = *(const v16bf*)&A1[(mt * 24 + kt) * 512 + lane * 16];
#pragma unroll
    for (int p = 0; p < 4; ++p) {
      v16bf b = *(const v16bf*)(W1b +
                 ((((kt * 16 + wave * 4 + p) * 32) + lane) << 4));
#pragma unroll
      for (int mt = 0; mt < 4; ++mt)
        acc[mt][p] = __builtin_amdgcn_wmma_f32_16x16x32_bf16(
            false, a[mt], false, b, (short)0, acc[mt][p], false, false);
    }
  }

  // ReLU + repack into layer-2 A-layout (K2 = 256)
#pragma unroll
  for (int p = 0; p < 4; ++p) {
    const int k   = (wave * 4 + p) * 16 + m;   // layer-2 K index
    const int kt2 = k >> 5, kk = k & 31;
    const int eidx    = ((kk >> 4) << 3) | (kk & 7);
    const int dl_base = ((kk >> 3) & 1) << 4;
#pragma unroll
    for (int mt = 0; mt < 4; ++mt) {
#pragma unroll
      for (int r = 0; r < 8; ++r) {
        float v = fmaxf(acc[mt][p][r], 0.0f);
        h1[(mt * 8 + kt2) * 512 + (r + half * 8 + dl_base) * 16 + eidx] =
            (__bf16)v;
      }
    }
  }
  __syncthreads();

  // ---------------- Layer 2: 256 -> 64 (wave owns N-tile = wave) ----------
  v8f acc2[4];                          // [mt]
  {
    float bv = b2[wave * 16 + m];
    v8f t;
#pragma unroll
    for (int r = 0; r < 8; ++r) t[r] = bv;
#pragma unroll
    for (int mt = 0; mt < 4; ++mt) acc2[mt] = t;
  }
#pragma unroll
  for (int kt = 0; kt < 8; ++kt) {
    v16bf b = *(const v16bf*)(W2b + ((((kt * 4 + wave) * 32) + lane) << 4));
#pragma unroll
    for (int mt = 0; mt < 4; ++mt) {
      v16bf a2 = *(const v16bf*)&h1[(mt * 8 + kt) * 512 + lane * 16];
      acc2[mt] = __builtin_amdgcn_wmma_f32_16x16x32_bf16(
          false, a2, false, b, (short)0, acc2[mt], false, false);
    }
  }
#pragma unroll
  for (int mt = 0; mt < 4; ++mt) {
#pragma unroll
    for (int r = 0; r < 8; ++r) {
      int me = mt * 16 + r + half * 8;
      h2[me * 64 + wave * 16 + m] = fmaxf(acc2[mt][r], 0.0f);
    }
  }
  __syncthreads();

  // ---------------- Layers 3/4: 64 -> 20 -> 1, segment accumulate ---------
  if (tid < BLK_EDGES) {
    float hrow[64];
#pragma unroll
    for (int q = 0; q < 16; ++q)
      *(float4*)&hrow[q * 4] = *(const float4*)&h2[tid * 64 + q * 4];
    float w = b4[0];
    for (int j = 0; j < 20; ++j) {
      float s = b3[j];
#pragma unroll 8
      for (int k = 0; k < 64; ++k) s += hrow[k] * W3[k * 20 + j];
      w += fmaxf(s, 0.0f) * W4[j];
    }
    int cj = edge_j[base_edge + tid];
    atomicAdd(&sums[cj], w);
    atomicAdd(&cnts[cj], 1.0f);
  }
}

__global__ void means_kernel(const float* __restrict__ sums,
                             const float* __restrict__ cnts,
                             float* __restrict__ means) {
  int c = blockIdx.x * blockDim.x + threadIdx.x;
  if (c < N_CLAUSES) means[c] = sums[c] / fmaxf(cnts[c], 1.0f);
}

__global__ void scatter_kernel(const int* __restrict__ edge_j,
                               const float* __restrict__ means,
                               float* __restrict__ out) {
  int e = blockIdx.x * blockDim.x + threadIdx.x;
  if (e < N_EDGES) out[e] = means[edge_j[e]];
}

extern "C" void kernel_launch(void* const* d_in, const int* in_sizes, int n_in,
                              void* d_out, int out_size, void* d_ws, size_t ws_size,
                              hipStream_t stream) {
  const float* l_embs = (const float*)d_in[0];
  const float* c_embs = (const float*)d_in[1];
  const int*   edge_i = (const int*)d_in[2];
  const int*   edge_j = (const int*)d_in[3];
  const float* W1 = (const float*)d_in[4];
  const float* b1 = (const float*)d_in[5];
  const float* W2 = (const float*)d_in[6];
  const float* b2 = (const float*)d_in[7];
  const float* W3 = (const float*)d_in[8];
  const float* b3 = (const float*)d_in[9];
  const float* W4 = (const float*)d_in[10];
  const float* b4 = (const float*)d_in[11];
  float* out = (float*)d_out;

  char* ws = (char*)d_ws;
  __bf16* W1b  = (__bf16*)(ws);
  __bf16* W2b  = (__bf16*)(ws + 393216);
  float* sums  = (float*)(ws + 425984);
  float* cnts  = (float*)(ws + 688128);
  float* means = (float*)(ws + 950272);

  zero_f32_kernel<<<(2 * N_CLAUSES + 255) / 256, 256, 0, stream>>>(
      sums, 2 * N_CLAUSES);
  prep_wmma_b_kernel<<<(K1 * N1 + 255) / 256, 256, 0, stream>>>(W1, W1b, K1, N1);
  prep_wmma_b_kernel<<<(K2 * N2 + 255) / 256, 256, 0, stream>>>(W2, W2b, K2, N2);

  edge_mlp_kernel<<<N_EDGES / BLK_EDGES, 128, LDS_TOTAL, stream>>>(
      l_embs, c_embs, edge_i, edge_j, W1b, b1, W2b, b2, W3, b3, W4, b4,
      sums, cnts);

  means_kernel<<<(N_CLAUSES + 255) / 256, 256, 0, stream>>>(sums, cnts, means);
  scatter_kernel<<<(N_EDGES + 255) / 256, 256, 0, stream>>>(edge_j, means, out);
}
